// EntityResolutionGNN_68453188764093
// MI455X (gfx1250) — matrix operand order repaired and checked by reading, not compiled, not run on hardware
//
#include <hip/hip_runtime.h>
#include <hip/hip_bf16.h>

// ---------------------------------------------------------------------------
// Problem constants (match reference)
// ---------------------------------------------------------------------------
#define N_ROWS 20000
#define N_TOK  200000
#define NEDGE  300000
#define FIN    300      // ROW_IN == TOK_IN == COL_IN
#define HID    256
#define EDIM   128
#define NHEAD  4
#define HD     64
#define NLAYER 3
#define NOUT   128

typedef __attribute__((ext_vector_type(16))) __bf16 v16bf;
typedef __attribute__((ext_vector_type(8)))  float  v8f;

struct alignas(16) U4 { unsigned x, y, z, w; };

union ABf {
  unsigned short s[16];
  U4 q[2];
  v16bf v;
};

// pack two f32 -> one dword of 2x bf16; let the backend pick the native
// convert (fptrunc f32->bf16), which can fuse into v_cvt_pk_bf16_f32.
__device__ __forceinline__ unsigned pk2(float a, float b) {
  union { __bf16 h[2]; unsigned u; } c;
  c.h[0] = (__bf16)a;
  c.h[1] = (__bf16)b;
  return c.u;
}

__device__ __forceinline__ float gelu_exact(float x) {
  return 0.5f * x * (1.0f + erff(x * 0.70710678118654752440f));
}

// monotone float <-> uint key (for atomic segment-max)
__device__ __forceinline__ unsigned fkey(float f) {
  unsigned u = __float_as_uint(f);
  return (u & 0x80000000u) ? ~u : (u | 0x80000000u);
}
__device__ __forceinline__ float funkey(unsigned k) {
  unsigned u = (k & 0x80000000u) ? (k & 0x7FFFFFFFu) : ~k;
  return __uint_as_float(u);
}
#define NEG_INF_KEY 0x007FFFFFu   /* fkey(-inf) */

// ---------------------------------------------------------------------------
// WMMA bf16 GEMM:  C[M,N] = act( A[M,K](lda) * W[K,N] + bias )
// N % 64 == 0, M % 16 == 0 (true for every GEMM here); K arbitrary.
// Block = 256 threads (8 waves) -> 128x64 tile. Double-buffered LDS staging:
// while each wave consumes bf16 fragments of step t (aligned ds_load_b128 ->
// 4x v_wmma_f32_16x16x32_bf16), the block cooperatively stages step t+1.
// Full/partial step selection is WAVE-UNIFORM (kk+32<=K, scalar branch only);
// the single partial step uses clamped-index + select (no EXEC toggling).
// B staging: waves 0-3 (scalar predicate via readfirstlane) stage (k,k+1)
// row pairs -> packed ds_store_b32.
// ---------------------------------------------------------------------------
#define MTILE 128
#define NTILE 64
#define ASTR  40    /* ushorts per A-row slot in LDS (80B: 16B-aligned, bank-spread) */
#define BSTR  40    /* ushorts per B-col slot in LDS */
#define ABUFSZ (MTILE * ASTR)
#define BBUFSZ (NTILE * BSTR)

template <int ACT>
__global__ __launch_bounds__(256)
void gemm_bf16_wmma(const float* __restrict__ A, int lda,
                    const float* __restrict__ W,
                    const float* __restrict__ bias,
                    float* __restrict__ C,
                    int M, int N, int K) {
  __shared__ unsigned short lA[2 * ABUFSZ];
  __shared__ unsigned short lB[2 * BBUFSZ];

  const int tid  = threadIdx.x;
  const int lane = tid & 31;
  const int wid  = tid >> 5;
  const int m0   = blockIdx.y * MTILE;
  const int n0   = blockIdx.x * NTILE;

  // --- staging assignments ---
  const int s_arow = tid >> 1;                 // 0..127
  const int s_ah   = (tid & 1) << 4;           // 0 / 16 (k offset within step)
  int grow = m0 + s_arow; if (grow >= M) grow = M - 1;
  const float* Arow = A + (long)grow * lda;

  const int s_bk0 = (tid >> 3) << 1;           // even k: 0,2,...,30 (waves 0-3)
  const int s_bn2 = (tid & 7) << 3;            // 0,8,...,56
  // wave-uniform predicate routed through an SGPR -> scalar branch, no EXEC ops
  const bool bstager = (__builtin_amdgcn_readfirstlane(wid) < 4);

  // --- fragment read coordinates ---
  const int f_arow = (wid << 4) + (lane & 15);         // wave row in tile
  const int f_akb  = (lane >> 4) << 4;                 // A: +0 / +16
  const int f_bkb  = (lane >> 4) << 4;                 // B: +0 / +16
  const int f_ncol = lane & 15;

  auto storeA = [&](unsigned short* sA, const float* f) {
    U4 u0, u1;
    u0.x = pk2(f[0], f[1]);   u0.y = pk2(f[2], f[3]);
    u0.z = pk2(f[4], f[5]);   u0.w = pk2(f[6], f[7]);
    u1.x = pk2(f[8], f[9]);   u1.y = pk2(f[10], f[11]);
    u1.z = pk2(f[12], f[13]); u1.w = pk2(f[14], f[15]);
    *(U4*)(sA + s_arow * ASTR + s_ah)     = u0;
    *(U4*)(sA + s_arow * ASTR + s_ah + 8) = u1;
  };
  // store a (k,k+1) pair for 8 columns: 8 packed dword stores
  auto storeB = [&](unsigned short* sB, const float4& a0, const float4& a1,
                    const float4& c0, const float4& c1) {
    unsigned short* q0 = sB + s_bn2 * BSTR + s_bk0;
    *(unsigned*)(q0 + 0 * BSTR) = pk2(a0.x, c0.x);
    *(unsigned*)(q0 + 1 * BSTR) = pk2(a0.y, c0.y);
    *(unsigned*)(q0 + 2 * BSTR) = pk2(a0.z, c0.z);
    *(unsigned*)(q0 + 3 * BSTR) = pk2(a0.w, c0.w);
    *(unsigned*)(q0 + 4 * BSTR) = pk2(a1.x, c1.x);
    *(unsigned*)(q0 + 5 * BSTR) = pk2(a1.y, c1.y);
    *(unsigned*)(q0 + 6 * BSTR) = pk2(a1.z, c1.z);
    *(unsigned*)(q0 + 7 * BSTR) = pk2(a1.w, c1.w);
  };

  // stage K-step t into LDS buffer `buf`
  auto stage = [&](int t, int buf) {
    unsigned short* sA = lA + buf * ABUFSZ;
    unsigned short* sB = lB + buf * BBUFSZ;
    const int kk = t << 5;
    if (kk + 32 <= K) {
      // ---------- full step: zero guards, pure b128 clauses ----------
      {
        const float4* ap = (const float4*)(Arow + kk + s_ah);
        float4 f0 = ap[0], f1 = ap[1], f2 = ap[2], f3 = ap[3];
        float f[16] = {f0.x, f0.y, f0.z, f0.w, f1.x, f1.y, f1.z, f1.w,
                       f2.x, f2.y, f2.z, f2.w, f3.x, f3.y, f3.z, f3.w};
        storeA(sA, f);
      }
      if (bstager) {                           // scalar branch (waves 0-3)
        const float* w0 = W + (long)(kk + s_bk0) * N + n0 + s_bn2;
        const float4* wp0 = (const float4*)w0;
        const float4* wp1 = (const float4*)(w0 + N);
        float4 a0 = wp0[0], a1 = wp0[1];
        float4 c0 = wp1[0], c1 = wp1[1];
        storeB(sB, a0, a1, c0, c1);
      }
    } else {
      // ---------- single partial step: branch-free clamp + select ----------
      {
        const int kb = kk + s_ah;
        float f[16];
#pragma unroll
        for (int j = 0; j < 16; ++j) {
          int k  = kb + j;
          int kc = k < K ? k : (K - 1);
          float x = Arow[kc];
          f[j] = (k < K) ? x : 0.0f;
        }
        storeA(sA, f);
      }
      if (bstager) {
        const int kg0 = kk + s_bk0, kg1 = kg0 + 1;
        const int kc0 = kg0 < K ? kg0 : (K - 1);
        const int kc1 = kg1 < K ? kg1 : (K - 1);
        const float4* wp0 = (const float4*)(W + (long)kc0 * N + n0 + s_bn2);
        const float4* wp1 = (const float4*)(W + (long)kc1 * N + n0 + s_bn2);
        const float z0 = (kg0 < K) ? 1.0f : 0.0f;
        const float z1 = (kg1 < K) ? 1.0f : 0.0f;
        float4 a0 = wp0[0], a1 = wp0[1];
        float4 c0 = wp1[0], c1 = wp1[1];
        a0.x *= z0; a0.y *= z0; a0.z *= z0; a0.w *= z0;
        a1.x *= z0; a1.y *= z0; a1.z *= z0; a1.w *= z0;
        c0.x *= z1; c0.y *= z1; c0.z *= z1; c0.w *= z1;
        c1.x *= z1; c1.y *= z1; c1.z *= z1; c1.w *= z1;
        storeB(sB, a0, a1, c0, c1);
      }
    }
  };

  v8f acc0 = {}, acc1 = {}, acc2 = {}, acc3 = {};
  const int nk = (K + 31) >> 5;

  stage(0, 0);                                  // prologue

  for (int t = 0; t < nk; ++t) {
    __syncthreads();                            // buffer (t&1) ready; other buffer free
    if (t + 1 < nk) stage(t + 1, (t + 1) & 1);  // overlap next-step staging with math

    const unsigned short* sA = lA + (t & 1) * ABUFSZ;
    const unsigned short* sB = lB + (t & 1) * BBUFSZ;

    ABf a;
    a.q[0] = *(const U4*)(sA + f_arow * ASTR + f_akb);        // K = akb+0..7
    a.q[1] = *(const U4*)(sA + f_arow * ASTR + f_akb + 8);    // K = akb+8..15

    ABf b0, b1, b2, b3;
    {
      const unsigned short* bp0 = sB + (f_ncol +  0) * BSTR + f_bkb;
      const unsigned short* bp1 = sB + (f_ncol + 16) * BSTR + f_bkb;
      const unsigned short* bp2 = sB + (f_ncol + 32) * BSTR + f_bkb;
      const unsigned short* bp3 = sB + (f_ncol + 48) * BSTR + f_bkb;
      b0.q[0] = *(const U4*)bp0;  b0.q[1] = *(const U4*)(bp0 + 8);
      b1.q[0] = *(const U4*)bp1;  b1.q[1] = *(const U4*)(bp1 + 8);
      b2.q[0] = *(const U4*)bp2;  b2.q[1] = *(const U4*)(bp2 + 8);
      b3.q[0] = *(const U4*)bp3;  b3.q[1] = *(const U4*)(bp3 + 8);
    }

    acc0 = __builtin_amdgcn_wmma_f32_16x16x32_bf16(false, a.v, false, b0.v, (short)0, acc0, false, false);
    acc1 = __builtin_amdgcn_wmma_f32_16x16x32_bf16(false, a.v, false, b1.v, (short)0, acc1, false, false);
    acc2 = __builtin_amdgcn_wmma_f32_16x16x32_bf16(false, a.v, false, b2.v, (short)0, acc2, false, false);
    acc3 = __builtin_amdgcn_wmma_f32_16x16x32_bf16(false, a.v, false, b3.v, (short)0, acc3, false, false);
  }

  // ---- epilogue ----
  const int rbase = m0 + (wid << 4) + ((lane >> 4) << 3);
  const int cbase = n0 + (lane & 15);
  float bv0 = 0.f, bv1 = 0.f, bv2 = 0.f, bv3 = 0.f;
  if (bias) {
    bv0 = bias[cbase];      bv1 = bias[cbase + 16];
    bv2 = bias[cbase + 32]; bv3 = bias[cbase + 48];
  }
#pragma unroll
  for (int g = 0; g < 8; ++g) {
    int r = rbase + g;
    if (r < M) {
      float v0 = acc0[g] + bv0, v1 = acc1[g] + bv1;
      float v2 = acc2[g] + bv2, v3 = acc3[g] + bv3;
      if (ACT == 1) {
        v0 = gelu_exact(v0); v1 = gelu_exact(v1);
        v2 = gelu_exact(v2); v3 = gelu_exact(v3);
      }
      float* cr = C + (long)r * N + cbase;
      cr[0] = v0; cr[16] = v1; cr[32] = v2; cr[48] = v3;
    }
  }
}

// ---------------------------------------------------------------------------
// LayerNorm(+GELU) over HID=256 features, one wave per row (float4 access).
// ---------------------------------------------------------------------------
__global__ __launch_bounds__(256)
void ln_gelu_kernel(const float* __restrict__ y,
                    const float* __restrict__ g, const float* __restrict__ b,
                    float* __restrict__ out, int ldo, int M) {
  const int lane = threadIdx.x & 31;
  const int row  = blockIdx.x * 8 + (threadIdx.x >> 5);
  if (row >= M) return;
  const int c0 = lane * 8;
  const float4* yr = (const float4*)(y + (long)row * HID + c0);
  float4 y0 = yr[0], y1 = yr[1];
  float v[8] = {y0.x, y0.y, y0.z, y0.w, y1.x, y1.y, y1.z, y1.w};
  float s = 0.f, ss = 0.f;
#pragma unroll
  for (int j = 0; j < 8; ++j) { s += v[j]; ss += v[j] * v[j]; }
#pragma unroll
  for (int m = 16; m >= 1; m >>= 1) { s += __shfl_xor(s, m, 32); ss += __shfl_xor(ss, m, 32); }
  float mu   = s * (1.0f / HID);
  float var  = ss * (1.0f / HID) - mu * mu;
  float rstd = rsqrtf(var + 1e-5f);
  const float4* gp = (const float4*)(g + c0);
  const float4* bp = (const float4*)(b + c0);
  float4 g0 = gp[0], g1 = gp[1], b0 = bp[0], b1 = bp[1];
  float gg[8] = {g0.x, g0.y, g0.z, g0.w, g1.x, g1.y, g1.z, g1.w};
  float bb[8] = {b0.x, b0.y, b0.z, b0.w, b1.x, b1.y, b1.z, b1.w};
  float o[8];
#pragma unroll
  for (int j = 0; j < 8; ++j) o[j] = gelu_exact((v[j] - mu) * rstd * gg[j] + bb[j]);
  float4* orow = (float4*)(out + (long)row * ldo + c0);
  orow[0] = make_float4(o[0], o[1], o[2], o[3]);
  orow[1] = make_float4(o[4], o[5], o[6], o[7]);
}

// x_new = LN(x_old + agg + skip); strided in/out (row_x lives in JK cat buffer)
__global__ __launch_bounds__(256)
void residual_ln_kernel(const float* __restrict__ xold, int ldx,
                        const float* __restrict__ agg,
                        const float* __restrict__ skip,
                        const float* __restrict__ g, const float* __restrict__ b,
                        float* __restrict__ out, int ldo, int M) {
  const int lane = threadIdx.x & 31;
  const int row  = blockIdx.x * 8 + (threadIdx.x >> 5);
  if (row >= M) return;
  const int c0 = lane * 8;
  const float4* xr = (const float4*)(xold + (long)row * ldx + c0);
  const float4* ar = (const float4*)(agg  + (long)row * HID + c0);
  const float4* sr = (const float4*)(skip + (long)row * HID + c0);
  float4 x0 = xr[0], x1 = xr[1], a0 = ar[0], a1 = ar[1], s0 = sr[0], s1 = sr[1];
  float v[8] = {x0.x + a0.x + s0.x, x0.y + a0.y + s0.y, x0.z + a0.z + s0.z, x0.w + a0.w + s0.w,
                x1.x + a1.x + s1.x, x1.y + a1.y + s1.y, x1.z + a1.z + s1.z, x1.w + a1.w + s1.w};
  float s = 0.f, ss = 0.f;
#pragma unroll
  for (int j = 0; j < 8; ++j) { s += v[j]; ss += v[j] * v[j]; }
#pragma unroll
  for (int m = 16; m >= 1; m >>= 1) { s += __shfl_xor(s, m, 32); ss += __shfl_xor(ss, m, 32); }
  float mu   = s * (1.0f / HID);
  float var  = ss * (1.0f / HID) - mu * mu;
  float rstd = rsqrtf(var + 1e-5f);
  const float4* gp = (const float4*)(g + c0);
  const float4* bp = (const float4*)(b + c0);
  float4 g0 = gp[0], g1 = gp[1], b0 = bp[0], b1 = bp[1];
  float gg[8] = {g0.x, g0.y, g0.z, g0.w, g1.x, g1.y, g1.z, g1.w};
  float bb[8] = {b0.x, b0.y, b0.z, b0.w, b1.x, b1.y, b1.z, b1.w};
  float o[8];
#pragma unroll
  for (int j = 0; j < 8; ++j) o[j] = (v[j] - mu) * rstd * gg[j] + bb[j];
  float4* orow = (float4*)(out + (long)row * ldo + c0);
  orow[0] = make_float4(o[0], o[1], o[2], o[3]);
  orow[1] = make_float4(o[4], o[5], o[6], o[7]);
}

// ---------------------------------------------------------------------------
// Edge attention kernels (one wave per edge; 8 dims/lane; head = lane>>3)
// ---------------------------------------------------------------------------
__global__ __launch_bounds__(256)
void edge_alpha_kernel(const float* __restrict__ q, const float* __restrict__ k,
                       const float* __restrict__ ep,
                       const int* __restrict__ src, const int* __restrict__ dst,
                       float* __restrict__ alpha, unsigned* __restrict__ mkey, int E) {
  const int lane = threadIdx.x & 31;
  const int e = blockIdx.x * 8 + (threadIdx.x >> 5);
  if (e >= E) return;
  const int d0 = lane * 8;
  const int sd = dst[e], ssrc = src[e];
  const float4* qr = (const float4*)(q  + (long)sd   * HID + d0);
  const float4* kr = (const float4*)(k  + (long)ssrc * HID + d0);
  const float4* er = (const float4*)(ep + (long)e    * HID + d0);
  float4 q0 = qr[0], q1 = qr[1], k0 = kr[0], k1 = kr[1], e0 = er[0], e1 = er[1];
  float p = q0.x * (k0.x + e0.x) + q0.y * (k0.y + e0.y)
          + q0.z * (k0.z + e0.z) + q0.w * (k0.w + e0.w)
          + q1.x * (k1.x + e1.x) + q1.y * (k1.y + e1.y)
          + q1.z * (k1.z + e1.z) + q1.w * (k1.w + e1.w);
  p += __shfl_xor(p, 1, 32); p += __shfl_xor(p, 2, 32); p += __shfl_xor(p, 4, 32);
  if ((lane & 7) == 0) {
    int h = lane >> 3;
    float a = p * 0.125f;                       // 1/sqrt(HD=64)
    alpha[e * NHEAD + h] = a;
    atomicMax(&mkey[sd * NHEAD + h], fkey(a));
  }
}

__global__ __launch_bounds__(256)
void edge_softmax_kernel(float* __restrict__ alpha, const unsigned* __restrict__ mkey,
                         float* __restrict__ ssum, const int* __restrict__ dst, int E4) {
  int i = blockIdx.x * 256 + threadIdx.x;
  if (i >= E4) return;
  int e = i >> 2, h = i & 3;
  int d = dst[e];
  float w = expf(alpha[i] - funkey(mkey[d * NHEAD + h]));
  alpha[i] = w;
  atomicAdd(&ssum[d * NHEAD + h], w);
}

__global__ __launch_bounds__(256)
void edge_agg_kernel(const float* __restrict__ v, const float* __restrict__ ep,
                     const float* __restrict__ alpha, const float* __restrict__ ssum,
                     const int* __restrict__ src, const int* __restrict__ dst,
                     float* __restrict__ agg, int E) {
  const int lane = threadIdx.x & 31;
  const int e = blockIdx.x * 8 + (threadIdx.x >> 5);
  if (e >= E) return;
  const int h = lane >> 3;
  const int d = dst[e], sidx = src[e];
  const float a = alpha[e * NHEAD + h] / (ssum[d * NHEAD + h] + 1e-16f);
  const int d0 = lane * 8;
  const float4* vr = (const float4*)(v  + (long)sidx * HID + d0);
  const float4* er = (const float4*)(ep + (long)e    * HID + d0);
  float4 v0 = vr[0], v1 = vr[1], e0 = er[0], e1 = er[1];
  float* ar = agg + (long)d * HID + d0;
  atomicAdd(&ar[0], (v0.x + e0.x) * a);
  atomicAdd(&ar[1], (v0.y + e0.y) * a);
  atomicAdd(&ar[2], (v0.z + e0.z) * a);
  atomicAdd(&ar[3], (v0.w + e0.w) * a);
  atomicAdd(&ar[4], (v1.x + e1.x) * a);
  atomicAdd(&ar[5], (v1.y + e1.y) * a);
  atomicAdd(&ar[6], (v1.z + e1.z) * a);
  atomicAdd(&ar[7], (v1.w + e1.w) * a);
}

// ---------------------------------------------------------------------------
// Misc
// ---------------------------------------------------------------------------
__global__ void fill_f32_kernel(float* p, float v, int n) {
  int i = blockIdx.x * 256 + threadIdx.x; if (i < n) p[i] = v;
}
__global__ void fill_u32_kernel(unsigned* p, unsigned v, int n) {
  int i = blockIdx.x * 256 + threadIdx.x; if (i < n) p[i] = v;
}

__global__ __launch_bounds__(256)
void rownorm_kernel(const float* __restrict__ h2, float* __restrict__ out, int M) {
  const int lane = threadIdx.x & 31;
  const int row  = blockIdx.x * 8 + (threadIdx.x >> 5);
  if (row >= M) return;
  const float4* hr = (const float4*)(h2 + (long)row * NOUT + lane * 4);
  float4 h = hr[0];
  float ss = h.x * h.x + h.y * h.y + h.z * h.z + h.w * h.w;
#pragma unroll
  for (int m = 16; m >= 1; m >>= 1) ss += __shfl_xor(ss, m, 32);
  float inv = 1.0f / fmaxf(sqrtf(ss), 1e-12f);
  float4* orow = (float4*)(out + (long)row * NOUT + lane * 4);
  orow[0] = make_float4(h.x * inv, h.y * inv, h.z * inv, h.w * inv);
}

// ---------------------------------------------------------------------------
// Host side
// ---------------------------------------------------------------------------
static inline int cdiv(long a, long b) { return (int)((a + b - 1) / b); }

static void launch_gemm(int act, const float* A, int lda, const float* W,
                        const float* bias, float* C, int M, int N, int K,
                        hipStream_t st) {
  dim3 grid(N / NTILE, cdiv(M, MTILE));
  if (act)
    gemm_bf16_wmma<1><<<grid, 256, 0, st>>>(A, lda, W, bias, C, M, N, K);
  else
    gemm_bf16_wmma<0><<<grid, 256, 0, st>>>(A, lda, W, bias, C, M, N, K);
}

struct Scratch {
  float *cat, *tok, *ea0, *ea1, *qb, *kb, *vb, *skipb, *epb, *alphab;
  unsigned* mkey;
  float *sb, *aggb, *jkob, *h1b, *h2b;
};

static void run_tconv(const Scratch& S,
                      const float* xsrc, int lds, int n_src,
                      const float* xdst, int ldd, int n_dst,
                      const float* ea, const int* src, const int* dst,
                      const float* Wq, const float* bq,
                      const float* Wk, const float* bk,
                      const float* Wv, const float* bv,
                      const float* We,
                      const float* Ws, const float* bs,
                      const float* lng, const float* lnb,
                      float* xout, int ldo, hipStream_t st) {
  launch_gemm(0, xdst, ldd, Wq, bq, S.qb,    n_dst, HID, HID, st);
  launch_gemm(0, xsrc, lds, Wk, bk, S.kb,    n_src, HID, HID, st);
  launch_gemm(0, xsrc, lds, Wv, bv, S.vb,    n_src, HID, HID, st);
  launch_gemm(0, xdst, ldd, Ws, bs, S.skipb, n_dst, HID, HID, st);
  launch_gemm(0, ea, EDIM, We, nullptr, S.epb, NEDGE, HID, EDIM, st);

  fill_u32_kernel<<<cdiv(n_dst * NHEAD, 256), 256, 0, st>>>(S.mkey, NEG_INF_KEY, n_dst * NHEAD);
  fill_f32_kernel<<<cdiv(n_dst * NHEAD, 256), 256, 0, st>>>(S.sb, 0.0f, n_dst * NHEAD);
  fill_f32_kernel<<<cdiv((long)n_dst * HID, 256), 256, 0, st>>>(S.aggb, 0.0f, n_dst * HID);

  int eb = cdiv(NEDGE, 8);
  edge_alpha_kernel<<<eb, 256, 0, st>>>(S.qb, S.kb, S.epb, src, dst, S.alphab, S.mkey, NEDGE);
  edge_softmax_kernel<<<cdiv(NEDGE * NHEAD, 256), 256, 0, st>>>(S.alphab, S.mkey, S.sb, dst, NEDGE * NHEAD);
  edge_agg_kernel<<<eb, 256, 0, st>>>(S.vb, S.epb, S.alphab, S.sb, src, dst, S.aggb, NEDGE);

  residual_ln_kernel<<<cdiv(n_dst, 8), 256, 0, st>>>(xdst, ldd, S.aggb, S.skipb, lng, lnb, xout, ldo, n_dst);
}

extern "C" void kernel_launch(void* const* d_in, const int* in_sizes, int n_in,
                              void* d_out, int out_size, void* d_ws, size_t ws_size,
                              hipStream_t stream) {
  (void)in_sizes; (void)n_in; (void)out_size; (void)ws_size;
  int i = 0;
  const float* row_feat   = (const float*)d_in[i++];
  const float* token_feat = (const float*)d_in[i++];
  const float* ea_t2r_in  = (const float*)d_in[i++];
  const float* ea_r2t_in  = (const float*)d_in[i++];
  const int*   src_t2r    = (const int*)d_in[i++];
  const int*   dst_t2r    = (const int*)d_in[i++];
  const int*   src_r2t    = (const int*)d_in[i++];
  const int*   dst_r2t    = (const int*)d_in[i++];
  const float* row_proj_W = (const float*)d_in[i++];
  const float* row_proj_b = (const float*)d_in[i++];
  const float* row_pln_g  = (const float*)d_in[i++];
  const float* row_pln_b  = (const float*)d_in[i++];
  const float* tok_proj_W = (const float*)d_in[i++];
  const float* tok_proj_b = (const float*)d_in[i++];
  const float* tok_pln_g  = (const float*)d_in[i++];
  const float* tok_pln_b  = (const float*)d_in[i++];
  const float* edge_proj_W = (const float*)d_in[i++];
  const float* edge_proj_b = (const float*)d_in[i++];
  const float* Wq = (const float*)d_in[i++];
  const float* bq = (const float*)d_in[i++];
  const float* Wk = (const float*)d_in[i++];
  const float* bk = (const float*)d_in[i++];
  const float* Wv = (const float*)d_in[i++];
  const float* bv = (const float*)d_in[i++];
  const float* We = (const float*)d_in[i++];
  const float* Wskip = (const float*)d_in[i++];
  const float* bskip = (const float*)d_in[i++];
  const float* ln_row_g = (const float*)d_in[i++];
  const float* ln_row_b = (const float*)d_in[i++];
  const float* ln_tok_g = (const float*)d_in[i++];
  const float* ln_tok_b = (const float*)d_in[i++];
  const float* jk_W = (const float*)d_in[i++];
  const float* jk_b = (const float*)d_in[i++];
  const float* head_W1 = (const float*)d_in[i++];
  const float* head_b1 = (const float*)d_in[i++];
  const float* head_W2 = (const float*)d_in[i++];
  const float* head_b2 = (const float*)d_in[i++];

  float* ws = (float*)d_ws;
  size_t off = 0;
  auto alloc = [&](size_t n) { float* p = ws + off; off += n; return p; };
  Scratch S;
  S.cat    = alloc((size_t)N_ROWS * HID * (NLAYER + 1));  // JK concat; row_x lives here
  S.tok    = alloc((size_t)N_TOK * HID);
  S.ea0    = alloc((size_t)NEDGE * EDIM);
  S.ea1    = alloc((size_t)NEDGE * EDIM);
  S.qb     = alloc((size_t)N_TOK * HID);
  S.kb     = alloc((size_t)N_TOK * HID);
  S.vb     = alloc((size_t)N_TOK * HID);
  S.skipb  = alloc((size_t)N_TOK * HID);
  S.epb    = alloc((size_t)NEDGE * HID);
  S.alphab = alloc((size_t)NEDGE * NHEAD);
  S.mkey   = (unsigned*)alloc((size_t)N_TOK * NHEAD);
  S.sb     = alloc((size_t)N_TOK * NHEAD);
  S.aggb   = alloc((size_t)N_TOK * HID);
  S.jkob   = alloc((size_t)N_ROWS * HID);
  S.h1b    = alloc((size_t)N_ROWS * HID);
  S.h2b    = alloc((size_t)N_ROWS * NOUT);

  const int CATLD = HID * (NLAYER + 1);   // 1024

  // --- input projections (GEMM -> LN+GELU) ---
  launch_gemm(0, row_feat, FIN, row_proj_W, row_proj_b, S.qb, N_ROWS, HID, FIN, stream);
  ln_gelu_kernel<<<cdiv(N_ROWS, 8), 256, 0, stream>>>(S.qb, row_pln_g, row_pln_b, S.cat, CATLD, N_ROWS);
  launch_gemm(0, token_feat, FIN, tok_proj_W, tok_proj_b, S.qb, N_TOK, HID, FIN, stream);
  ln_gelu_kernel<<<cdiv(N_TOK, 8), 256, 0, stream>>>(S.qb, tok_pln_g, tok_pln_b, S.tok, HID, N_TOK);

  // --- edge projections (GEMM + fused GELU) ---
  launch_gemm(1, ea_t2r_in, FIN, edge_proj_W, edge_proj_b, S.ea0, NEDGE, EDIM, FIN, stream);
  launch_gemm(1, ea_r2t_in, FIN, edge_proj_W, edge_proj_b, S.ea1, NEDGE, EDIM, FIN, stream);

  // --- conv layers ---
  for (int l = 0; l < NLAYER; ++l) {
    const float* rows_old = S.cat + (size_t)l * HID;
    float*       rows_new = S.cat + (size_t)(l + 1) * HID;
    size_t p2 = (size_t)(l * 2) * HID * HID;       // weight offset dir 0
    size_t b2 = (size_t)(l * 2) * HID;
    size_t e2 = (size_t)(l * 2) * EDIM * HID;
    // dir 0: token -> row
    run_tconv(S, S.tok, HID, N_TOK, rows_old, CATLD, N_ROWS,
              S.ea0, src_t2r, dst_t2r,
              Wq + p2, bq + b2, Wk + p2, bk + b2, Wv + p2, bv + b2,
              We + e2, Wskip + p2, bskip + b2,
              ln_row_g + (size_t)l * HID, ln_row_b + (size_t)l * HID,
              rows_new, CATLD, stream);
    size_t p2b = p2 + (size_t)HID * HID;
    size_t b2b = b2 + HID;
    size_t e2b = e2 + (size_t)EDIM * HID;
    // dir 1: row -> token (uses updated rows, in-place token LN)
    run_tconv(S, rows_new, CATLD, N_ROWS, S.tok, HID, N_TOK,
              S.ea1, src_r2t, dst_r2t,
              Wq + p2b, bq + b2b, Wk + p2b, bk + b2b, Wv + p2b, bv + b2b,
              We + e2b, Wskip + p2b, bskip + b2b,
              ln_tok_g + (size_t)l * HID, ln_tok_b + (size_t)l * HID,
              S.tok, HID, stream);
  }

  // --- jumping knowledge + head ---
  launch_gemm(0, S.cat, CATLD, jk_W, jk_b, S.jkob, N_ROWS, HID, CATLD, stream);
  launch_gemm(1, S.jkob, HID, head_W1, head_b1, S.h1b, N_ROWS, HID, HID, stream);
  launch_gemm(0, S.h1b, HID, head_W2, head_b2, S.h2b, N_ROWS, NOUT, HID, stream);
  rownorm_kernel<<<cdiv(N_ROWS, 8), 256, 0, stream>>>(S.h2b, (float*)d_out, N_ROWS);
}
